// Fembbeding_block_76407468196147
// MI455X (gfx1250) — compile-verified, matrix operand order checked by reading, not run on hardware
//
#include <hip/hip_runtime.h>

// ---------------------------------------------------------------------------
// MI455X (gfx1250) fused implementation, round 3.
//  - f16 WMMA (v_wmma_f32_16x16x32_f16), fp32 accumulate: memory-bound regime
//    (~122 MB ideal HBM traffic @23.3TB/s ~ 5.3us vs ~25 GFLOP).
//  - Async global->LDS weight staging (ASYNCcnt) double-buffered so the freq
//    weights stream in during the time-branch WMMAs.  Builtin signature per
//    compiler diagnostic: (AS1 int4* src, AS3 int4* dst, imm, imm).
//  - All LDS WMMA-fragment pitches padded (136 / 264 halves) so each wave's
//    16 b128 fragment loads hit distinct banks (4-bank stride per lane).
// ---------------------------------------------------------------------------

#define TS    128
#define STEP  64
#define NSEG  31
#define FD    128
#define NCH   12
#define TLEN  2048
#define LNSZ  (NSEG * FD)   // 3968
#define PW    136           // padded LDS pitch (halves): weights & A matrix
#define PC    264           // padded LDS pitch (halves): cat matrix

typedef _Float16 half_t;
typedef __attribute__((ext_vector_type(16))) _Float16 v16h;
typedef __attribute__((ext_vector_type(8)))  float    v8f;
typedef __attribute__((ext_vector_type(4)))  float    f4;
typedef __attribute__((ext_vector_type(4)))  int      i4v;

// ---- async global->LDS copy path (CDNA5), with synchronous fallback --------
#if defined(__has_builtin)
#if __has_builtin(__builtin_amdgcn_global_load_async_to_lds_b128)
#define USE_ASYNC 1
#endif
#endif
#ifndef USE_ASYNC
#define USE_ASYNC 0
#endif

#if USE_ASYNC
typedef __attribute__((address_space(1))) i4v i4v_g;   // global (AS1) int4
typedef __attribute__((address_space(3))) i4v i4v_l;   // LDS    (AS3) int4
#define ASYNC_CP16(dst, src)                                                   \
  __builtin_amdgcn_global_load_async_to_lds_b128((i4v_g*)(src), (i4v_l*)(dst), \
                                                 0, 0)
#define ASYNC_STR2(x) #x
#if __has_builtin(__builtin_amdgcn_s_wait_asynccnt)
#define ASYNC_WAIT(n) __builtin_amdgcn_s_wait_asynccnt(n)
#else
#define ASYNC_WAIT(n) asm volatile("s_wait_asynccnt " ASYNC_STR2(n) ::: "memory")
#endif
#else
#define ASYNC_CP16(dst, src) (*(f4*)(dst) = *(const f4*)(src))
#define ASYNC_WAIT(n) ((void)0)
#endif

union V16 { v16h v; f4 f[2]; };

// f16 WMMA fragment for this lane: 8 consecutive halves at p (K kb..kb+7) and
// 8 more at p+16 halves (K kb+16..kb+23).  Identical mapping for A and B ops.
__device__ __forceinline__ v16h ldfrag(const half_t* p) {
  V16 u;
  u.f[0] = *(const f4*)(p);
  u.f[1] = *(const f4*)(p + 16);
  return u.v;
}

__device__ __forceinline__ v8f wmma16(v16h a, v16h b, v8f c) {
  return __builtin_amdgcn_wmma_f32_16x16x32_f16(false, a, false, b,
                                                (short)0, c, false, false);
}

// ---- precompute kernel 1: T_w -> f16 transposed [c][d][t]; fus_w -> f16 ----
__global__ void k_convert(const float* __restrict__ T_w,
                          const float* __restrict__ fus_w,
                          half_t* __restrict__ twt,
                          half_t* __restrict__ fusw) {
  int i = blockIdx.x * 256 + threadIdx.x;
  if (i < NCH * TS * FD) {
    int c = i >> 14;
    int r = i & 16383;
    int t = r >> 7;
    int d = r & 127;
    twt[(c * FD + d) * TS + t] = (half_t)T_w[i];
  }
  if (i < 2 * FD * 2 * FD)
    fusw[i] = (half_t)fus_w[i];   // [j][i] row-major == N-major B layout
}

// ---- precompute kernel 2: fold DFT into F_w -> F_eff f16 [c][d][t] ---------
__global__ void k_fft_fold(const float* __restrict__ F_w,
                           half_t* __restrict__ fwt) {
  __shared__ float cs[TS];
  __shared__ float sn[TS];
  const int c = blockIdx.x >> 7;
  const int t = blockIdx.x & 127;
  const int d = threadIdx.x;      // 128 threads
  {
    int p = (d * t) & 127;                       // exact phase index
    float a = (float)p * 0.04908738521234052f;   // 2*pi/128
    cs[d] = cosf(a);
    sn[d] = sinf(a);
  }
  __syncthreads();
  const float* fw = F_w + (size_t)c * (2 * TS) * FD + d;
  float acc = 0.f;
#pragma unroll 4
  for (int k = 0; k < TS; ++k)
    acc = fmaf(cs[k], fw[k * FD], fmaf(-sn[k], fw[(TS + k) * FD], acc));
  fwt[((size_t)c * FD + d) * TS + t] = (half_t)acc;
}

// ---- main fused kernel: one 256-thread block (8 wave32) per (b, c) ---------
__global__ __launch_bounds__(256)
void k_main(const float* __restrict__ x,
            const half_t* __restrict__ twt, const float* __restrict__ T_b,
            const half_t* __restrict__ fwt, const float* __restrict__ F_b,
            const float* __restrict__ g_t, const float* __restrict__ b_t,
            const float* __restrict__ g_f, const float* __restrict__ b_f,
            const half_t* __restrict__ fusw, const float* __restrict__ fus_b,
            float* __restrict__ out) {
  union UB0 {                          // 34 KB: Wt tile, later fp32 out staging
    half_t w[FD * PW];
    float  o[FD * PW / 2];             // 8704 floats >= 31*256
  };
  __shared__ __align__(16) UB0    ub0;
  __shared__ __align__(16) half_t wfb[FD * PW];   // Wf tile (34 KB)
  __shared__ __align__(16) half_t amat[32 * PW];  // windowed A, f16 (8.5 KB)
  __shared__ __align__(16) half_t cat[32 * PC];   // LN output, f16 (16.5 KB)
  __shared__ __align__(16) f4     red[256];       // 4 KB reduction

  const int c    = blockIdx.x;
  const int b    = blockIdx.y;
  const int tid  = threadIdx.x;
  const int lane = tid & 31;
  const int wv   = tid >> 5;           // wave 0..7
  const int lr   = lane & 15;
  const int hi   = lane >> 4;
  const int koff = hi * 8;             // fragment K offset
  const int hrow = hi * 8;             // C/D M offset

  // ---- issue BOTH weight tiles as async global->LDS copies (8+8 per wave) --
  {
    const half_t* gw0 = twt + (size_t)c * FD * TS;
    const half_t* gw1 = fwt + (size_t)c * FD * TS;
    for (int i = tid; i < FD * TS / 8; i += 256) {   // 2048 16B chunks
      int row = i >> 4, cc = (i & 15) * 8;
      ASYNC_CP16(&ub0.w[row * PW + cc], gw0 + row * TS + cc);
    }
    for (int i = tid; i < FD * TS / 8; i += 256) {
      int row = i >> 4, cc = (i & 15) * 8;
      ASYNC_CP16(&wfb[row * PW + cc], gw1 + row * TS + cc);
    }
  }
  // warm L2 with this wave's fusion-weight columns (global_prefetch_b8)
  __builtin_prefetch(fusw + (size_t)(32 * wv + lane) * 256, 0, 3);

  // ---- build windowed A matrix (32 x 128, pitch PW) while copies fly -------
  {
    const float* xp = x + (size_t)b * TLEN * NCH + c;
    for (int t = tid; t < TLEN; t += 256) {
      half_t h = (half_t)xp[t * NCH];
      int s1 = t >> 6, col = t & 63;       // window s covers t in [64s,64s+128)
      amat[s1 * PW + col] = h;
      if (s1) amat[(s1 - 1) * PW + col + 64] = h;
    }
    if (tid < 8) {                          // zero tail of padding row 31
      f4 z = {};
      *(f4*)&amat[31 * PW + 64 + tid * 8] = z;
    }
  }
  ASYNC_WAIT(8);        // first 8 (Wt) done; Wf may still be in flight
  __syncthreads();

  // ---- stage 1, time branch (this wave: N tile = 16*wv) --------------------
  const int n0 = wv * 16;
  v8f at0 = {}, at1 = {};
  for (int k0 = 0; k0 < TS; k0 += 32) {
    v16h a0 = ldfrag(amat + PW * lr + k0 + koff);
    v16h a1 = ldfrag(amat + PW * (16 + lr) + k0 + koff);
    v16h bt = ldfrag(ub0.w + PW * (n0 + lr) + k0 + koff);
    at0 = wmma16(a0, bt, at0);
    at1 = wmma16(a1, bt, at1);
  }
  ASYNC_WAIT(0);        // freq weights landed (overlapped with the loop above)
  __syncthreads();

  // ---- stage 1, freq branch ------------------------------------------------
  v8f af0 = {}, af1 = {};
  for (int k0 = 0; k0 < TS; k0 += 32) {
    v16h a0 = ldfrag(amat + PW * lr + k0 + koff);
    v16h a1 = ldfrag(amat + PW * (16 + lr) + k0 + koff);
    v16h bf = ldfrag(wfb + PW * (n0 + lr) + k0 + koff);
    af0 = wmma16(a0, bf, af0);
    af1 = wmma16(a1, bf, af1);
  }

  // ---- bias + LayerNorm statistics over 31*128 values ----------------------
  const int   d  = n0 + lr;
  const float tb = T_b[c * FD + d];
  const float fb = F_b[c * FD + d];
  float s_t = 0.f, q_t = 0.f, s_f = 0.f, q_f = 0.f;
#pragma unroll
  for (int r = 0; r < 8; ++r) {
    at0[r] += tb; af0[r] += fb; at1[r] += tb; af1[r] += fb;
    float vt = at0[r], vf = af0[r];
    s_t += vt; q_t += vt * vt; s_f += vf; q_f += vf * vf;
    if (16 + r + hrow < NSEG) {          // exclude padding row 31
      vt = at1[r]; vf = af1[r];
      s_t += vt; q_t += vt * vt; s_f += vf; q_f += vf * vf;
    }
  }
  {
    f4 pv = {s_t, q_t, s_f, q_f};
    red[tid] = pv;
  }
  __syncthreads();
  for (int s = 128; s > 0; s >>= 1) {
    if (tid < s) red[tid] += red[tid + s];
    __syncthreads();
  }
  const float inv  = 1.0f / (float)LNSZ;
  f4 tot = red[0];
  const float mu_t = tot.x * inv, mu_f = tot.z * inv;
  const float rs_t = rsqrtf(tot.y * inv - mu_t * mu_t + 1e-5f);
  const float rs_f = rsqrtf(tot.w * inv - mu_f * mu_f + 1e-5f);

  // ---- normalize + affine + ReLU -> cat (f16, pitch PC) --------------------
#pragma unroll
  for (int r = 0; r < 8; ++r) {
    {
      int s  = r + hrow;                 // 0..15, always valid
      int gi = s * FD + d;
      float vt = fmaxf((at0[r] - mu_t) * rs_t * g_t[gi] + b_t[gi], 0.f);
      float vf = fmaxf((af0[r] - mu_f) * rs_f * g_f[gi] + b_f[gi], 0.f);
      cat[s * PC + d]      = (half_t)vt;
      cat[s * PC + FD + d] = (half_t)vf;
    }
    {
      int s = 16 + r + hrow;             // 16..31, row 31 padded to zero
      float vt = 0.f, vf = 0.f;
      if (s < NSEG) {
        int gi = s * FD + d;
        vt = fmaxf((at1[r] - mu_t) * rs_t * g_t[gi] + b_t[gi], 0.f);
        vf = fmaxf((af1[r] - mu_f) * rs_f * g_f[gi] + b_f[gi], 0.f);
      }
      cat[s * PC + d]      = (half_t)vt;
      cat[s * PC + FD + d] = (half_t)vf;
    }
  }
  __syncthreads();

  // ---- fusion GEMM: cat(32x256) @ fus_w.T  (B frags from L2-resident f16) --
#pragma unroll
  for (int nt = 0; nt < 2; ++nt) {
    const int col = 32 * wv + 16 * nt + lr;        // output column
    v8f c0 = {}, c1 = {};
    for (int k0 = 0; k0 < 2 * FD; k0 += 32) {
      v16h bfr = ldfrag(fusw + (size_t)col * 256 + k0 + koff);
      v16h a0  = ldfrag(cat + PC * lr + k0 + koff);
      v16h a1  = ldfrag(cat + PC * (16 + lr) + k0 + koff);
      c0 = wmma16(a0, bfr, c0);
      c1 = wmma16(a1, bfr, c1);
    }
    const float bias = fus_b[col];
#pragma unroll
    for (int r = 0; r < 8; ++r) {
      ub0.o[(r + hrow) * 256 + col] = c0[r] + bias;
      int s = 16 + r + hrow;
      if (s < NSEG) ub0.o[s * 256 + col] = c1[r] + bias;
    }
  }
  __syncthreads();

  // ---- coalesced fp32 store of the 31x256 block ----------------------------
  {
    float* op = out + ((size_t)b * (NCH * NSEG) + (size_t)c * NSEG) * 256;
    const f4* src = (const f4*)ub0.o;
    f4* dst = (f4*)op;
    for (int i = tid; i < NSEG * 256 / 4; i += 256) dst[i] = src[i];
  }
}

extern "C" void kernel_launch(void* const* d_in, const int* in_sizes, int n_in,
                              void* d_out, int out_size, void* d_ws, size_t ws_size,
                              hipStream_t stream) {
  const float* x     = (const float*)d_in[0];
  const float* T_w   = (const float*)d_in[1];
  const float* T_b   = (const float*)d_in[2];
  const float* F_w   = (const float*)d_in[3];
  const float* F_b   = (const float*)d_in[4];
  const float* ln_tg = (const float*)d_in[5];
  const float* ln_tb = (const float*)d_in[6];
  const float* ln_fg = (const float*)d_in[7];
  const float* ln_fb = (const float*)d_in[8];
  const float* fus_w = (const float*)d_in[9];
  const float* fus_b = (const float*)d_in[10];
  float* out = (float*)d_out;

  // workspace: f16 T_w^T | f16 F_eff^T | f16 fus_w   (~0.9 MB total)
  half_t* twt  = (half_t*)d_ws;
  half_t* fwt  = twt + (size_t)NCH * FD * TS;
  half_t* fusw = fwt + (size_t)NCH * FD * TS;

  const int B = in_sizes[0] / (TLEN * NCH);   // 256

  k_convert<<<dim3((NCH * TS * FD + 255) / 256), dim3(256), 0, stream>>>(
      T_w, fus_w, twt, fusw);
  k_fft_fold<<<dim3(NCH * TS), dim3(TS), 0, stream>>>(F_w, fwt);
  k_main<<<dim3(NCH, B), dim3(256), 0, stream>>>(
      x, twt, T_b, fwt, F_b, ln_tg, ln_tb, ln_fg, ln_fb, fusw, fus_b, out);
}
